// DecoderLayer_62105227100796
// MI455X (gfx1250) — compile-verified, matrix operand order checked by reading, not run on hardware
//
#include <hip/hip_runtime.h>
#include <stdint.h>

#define B_     2
#define S_     2048
#define SE_    2048
#define D_     512
#define H_     8
#define DEPTH_ 64
#define DFF_   2048

typedef __attribute__((ext_vector_type(16))) __bf16 v16bf;
typedef __attribute__((ext_vector_type(8)))  float  v8f;

__device__ inline unsigned short f32_to_bf16(float f) {
  union { float f; unsigned int u; } c; c.f = f;
  unsigned int u = c.u;
  unsigned int r = 0x7FFFu + ((u >> 16) & 1u);   // round-to-nearest-even
  return (unsigned short)((u + r) >> 16);
}

__device__ inline v8f wmma_bf16(v16bf a, v16bf b, v8f c) {
  return __builtin_amdgcn_wmma_f32_16x16x32_bf16(false, a, false, b, (short)0, c,
                                                 false, false);
}

// CDNA5 async memory->LDS DMA (ISA ch.10 / 15.18.3, tracked by ASYNCcnt).
// VDST VGPR holds the LDS byte address (low 32 bits of a generic LDS pointer
// are the pure LDS offset per the aperture rules); VADDR holds the global addr.
__device__ inline void async_ld_b128(void* lds, const void* gaddr) {
  unsigned int loff = (unsigned int)(uintptr_t)lds;
  asm volatile("global_load_async_to_lds_b128 %0, %1, off"
               :: "v"(loff), "v"(gaddr) : "memory");
}
__device__ inline void wait_async() {
  asm volatile("s_wait_asynccnt 0x0" ::: "memory");
}

__device__ inline float hmax16(float v) {
  v = fmaxf(v, __shfl_xor(v, 1, 32));
  v = fmaxf(v, __shfl_xor(v, 2, 32));
  v = fmaxf(v, __shfl_xor(v, 4, 32));
  v = fmaxf(v, __shfl_xor(v, 8, 32));
  return v;
}
__device__ inline float hsum16(float v) {
  v += __shfl_xor(v, 1, 32);
  v += __shfl_xor(v, 2, 32);
  v += __shfl_xor(v, 4, 32);
  v += __shfl_xor(v, 8, 32);
  return v;
}

// ---------------------------------------------------------------- f32 -> bf16
__global__ void cvt_f32_to_bf16(const float* __restrict__ in,
                                unsigned short* __restrict__ out, int n) {
  int i = blockIdx.x * blockDim.x + threadIdx.x;
  int stride = gridDim.x * blockDim.x;
  for (; i < n; i += stride) out[i] = f32_to_bf16(in[i]);
}

// ------------------------------------------- weight f32[K,N] -> bf16 Wt[N,K]
__global__ void cvt_transpose_bf16(const float* __restrict__ in,
                                   unsigned short* __restrict__ out,
                                   int K, int N) {
  __shared__ float t[32][33];
  const int n0 = blockIdx.x * 32;
  const int k0 = blockIdx.y * 32;
  const int tx = threadIdx.x & 31;
  const int ty = threadIdx.x >> 5;   // 0..7
#pragma unroll
  for (int i = 0; i < 32; i += 8)
    t[ty + i][tx] = in[(size_t)(k0 + ty + i) * N + n0 + tx];
  __syncthreads();
#pragma unroll
  for (int i = 0; i < 32; i += 8)
    out[(size_t)(n0 + ty + i) * K + k0 + tx] = f32_to_bf16(t[tx][ty + i]);
}

// ----------------------------------------------------------------- WMMA GEMM
// C[M,N] = act(A[M,K] @ Wt[N,K]^T + bias).  MODE: 0=f32 out, 1=bf16, 2=bf16+ReLU.
// 128 threads = 4 waves, each computes 64x64 (4x4 WMMA tiles); async-DMA
// double-buffered 128x32 tiles of A and Wt.
#define BM 128
#define BN 128
#define BK 32
#define LDT_P 48   // 32 + 16 pad; 96B row stride keeps 32B alignment

template <int MODE>
__launch_bounds__(128)
__global__ void gemm_bf16_wmma(const unsigned short* __restrict__ A,
                               const unsigned short* __restrict__ Wt,
                               const float* __restrict__ bias,
                               float* __restrict__ Cf,
                               unsigned short* __restrict__ Cb,
                               int M, int N, int K) {
  __shared__ __align__(32) unsigned short sA[2][BM * LDT_P];
  __shared__ __align__(32) unsigned short sB[2][BN * LDT_P];  // [n][k]

  const int tid  = threadIdx.x;
  const int lane = tid & 31;
  const int l15  = lane & 15;
  const int wid  = tid >> 5;   // 0..3
  const int wm   = wid >> 1;   // 2 waves along M (64 rows each)
  const int wn   = wid & 1;    // 2 waves along N (64 cols each)
  const int bm0  = blockIdx.y * BM;
  const int bn0  = blockIdx.x * BN;
  const int half16 = (lane >> 4) * 16;

  v8f acc[4][4];
#pragma unroll
  for (int i = 0; i < 4; ++i)
#pragma unroll
    for (int j = 0; j < 4; ++j)
#pragma unroll
      for (int r = 0; r < 8; ++r) acc[i][j][r] = 0.0f;

  auto issue_tile = [&](int buf, int k0) {
    unsigned short* dA = &sA[buf][0];
    unsigned short* dB = &sB[buf][0];
#pragma unroll
    for (int it = 0; it < 4; ++it) {
      int c   = tid + it * 128;        // 0..511
      int row = c >> 2;
      int col = (c & 3) * 8;
      async_ld_b128(&dA[row * LDT_P + col],
                    &A[(size_t)(bm0 + row) * K + k0 + col]);
      async_ld_b128(&dB[row * LDT_P + col],
                    &Wt[(size_t)(bn0 + row) * K + k0 + col]);
    }
  };

  // prologue: tile 0 into buffer 0
  issue_tile(0, 0);
  wait_async();
  __syncthreads();

  int buf = 0;
  for (int k0 = 0; k0 < K; k0 += BK) {
    const int nxt = k0 + BK;
    if (nxt < K) issue_tile(buf ^ 1, nxt);   // prefetch next tile (async DMA)

    const unsigned short* cA = &sA[buf][0];
    const unsigned short* cB = &sB[buf][0];
    v16bf aF[4], bF[4];
#pragma unroll
    for (int i = 0; i < 4; ++i)
      aF[i] = *(const v16bf*)(&cA[(wm * 64 + i * 16 + l15) * LDT_P + half16]);
#pragma unroll
    for (int j = 0; j < 4; ++j)
      bF[j] = *(const v16bf*)(&cB[(wn * 64 + j * 16 + l15) * LDT_P + half16]);
#pragma unroll
    for (int i = 0; i < 4; ++i)
#pragma unroll
      for (int j = 0; j < 4; ++j)
        acc[i][j] = wmma_bf16(aF[i], bF[j], acc[i][j]);

    wait_async();      // own prefetch done
    __syncthreads();   // all waves: prefetch visible + done reading old buffer
    buf ^= 1;
  }

  const int rowAdd = (lane >> 4) * 8;
#pragma unroll
  for (int i = 0; i < 4; ++i) {
#pragma unroll
    for (int j = 0; j < 4; ++j) {
      const int col = bn0 + wn * 64 + j * 16 + l15;
      const float bv = bias[col];
      size_t off = (size_t)(bm0 + wm * 64 + i * 16 + rowAdd) * N + col;
#pragma unroll
      for (int r = 0; r < 8; ++r) {
        float v = acc[i][j][r] + bv;
        if (MODE == 2) v = fmaxf(v, 0.0f);
        if (MODE == 0) Cf[off] = v;
        else           Cb[off] = f32_to_bf16(v);
        off += (size_t)N;
      }
    }
  }
}

// --------------------------------------------------------- flash attention
// grid: (Sq/64, H, B), block: 128 (4 waves). Each wave owns 16 q-rows.
// K tiles via async DMA, V tiles transposed via ds scatter; both ping-ponged
// so tile k+1 streams in while tile k is consumed by WMMA + softmax.
#define ATT_LDS 80   // 64 + 16 pad

__launch_bounds__(128)
__global__ void flash_attn_wmma(const unsigned short* __restrict__ Q,
                                const unsigned short* __restrict__ Kb,
                                const unsigned short* __restrict__ Vb,
                                const float* __restrict__ pad,  // [B,Skv]
                                unsigned short* __restrict__ O,
                                int Sq, int Skv, int causal, int usepad) {
  __shared__ __align__(32) unsigned short sQ[64 * ATT_LDS];
  __shared__ __align__(32) unsigned short sK[2][64 * ATT_LDS];
  __shared__ __align__(32) unsigned short sVt[2][DEPTH_ * ATT_LDS];  // [d][key]
  __shared__ __align__(32) unsigned short sP[4 * 16 * 64];           // per-wave P

  const int tid  = threadIdx.x;
  const int lane = tid & 31;
  const int l15  = lane & 15;
  const int wave = tid >> 5;
  const int half16 = (lane >> 4) * 16;
  const int rowAdd = (lane >> 4) * 8;
  const int q0 = blockIdx.x * 64;
  const int hh = blockIdx.y;
  const int bb = blockIdx.z;
  const int wbase = wave * 16 * 64;
  const float scale = 0.125f;  // rsqrt(64)

  auto loadKV = [&](int buf, int k0) {
    unsigned short* dK  = &sK[buf][0];
    unsigned short* dVt = &sVt[buf][0];
#pragma unroll
    for (int it = 0; it < 4; ++it) {
      int c   = tid + it * 128;
      int row = c >> 3;
      int col = (c & 7) * 8;
      async_ld_b128(&dK[row * ATT_LDS + col],
                    &Kb[(size_t)(bb * Skv + k0 + row) * D_ + hh * DEPTH_ + col]);
      uint4 w = *(const uint4*)(&Vb[(size_t)(bb * Skv + k0 + row) * D_ + hh * DEPTH_ + col]);
      unsigned int ws4[4] = {w.x, w.y, w.z, w.w};
#pragma unroll
      for (int e = 0; e < 4; ++e) {
        dVt[(col + 2 * e    ) * ATT_LDS + row] = (unsigned short)(ws4[e] & 0xffffu);
        dVt[(col + 2 * e + 1) * ATT_LDS + row] = (unsigned short)(ws4[e] >> 16);
      }
    }
  };

  // Q tile (64x64, async) + first K/V tile
#pragma unroll
  for (int it = 0; it < 4; ++it) {
    int c   = tid + it * 128;
    int row = c >> 3;
    int col = (c & 7) * 8;
    async_ld_b128(&sQ[row * ATT_LDS + col],
                  &Q[(size_t)(bb * Sq + q0 + row) * D_ + hh * DEPTH_ + col]);
  }
  loadKV(0, 0);
  wait_async();
  __syncthreads();

  float mrow[8], lrow[8];
  v8f oacc[4];
#pragma unroll
  for (int r = 0; r < 8; ++r) { mrow[r] = -1e30f; lrow[r] = 0.0f; }
#pragma unroll
  for (int j = 0; j < 4; ++j)
#pragma unroll
    for (int r = 0; r < 8; ++r) oacc[j][r] = 0.0f;

  const int kvEnd = causal ? (q0 + 64) : Skv;
  int buf = 0;
  for (int k0 = 0; k0 < kvEnd; k0 += 64) {
    const int nxt = k0 + 64;
    if (nxt < kvEnd) loadKV(buf ^ 1, nxt);   // prefetch next K/V tile

    const unsigned short* cK  = &sK[buf][0];
    const unsigned short* cVt = &sVt[buf][0];

    // S = Q @ K^T  (wave: 16 q-rows x 64 keys)
    v8f sacc[4];
#pragma unroll
    for (int j = 0; j < 4; ++j)
#pragma unroll
      for (int r = 0; r < 8; ++r) sacc[j][r] = 0.0f;
#pragma unroll
    for (int kk = 0; kk < 2; ++kk) {
      v16bf aq = *(const v16bf*)(&sQ[(wave * 16 + l15) * ATT_LDS + kk * 32 + half16]);
#pragma unroll
      for (int j = 0; j < 4; ++j) {
        v16bf bk = *(const v16bf*)(&cK[(j * 16 + l15) * ATT_LDS + kk * 32 + half16]);
        sacc[j] = wmma_bf16(aq, bk, sacc[j]);
      }
    }

    // scale + masks
    float sv[4][8];
    const int diag = causal && (k0 + 64 > q0);
#pragma unroll
    for (int j = 0; j < 4; ++j) {
      float pb = usepad ? pad[(size_t)bb * Skv + k0 + j * 16 + l15] * (-1e9f) : 0.0f;
#pragma unroll
      for (int r = 0; r < 8; ++r) {
        float s = sacc[j][r] * scale + pb;
        if (diag) {
          int qq = wave * 16 + r + rowAdd;
          int kc = j * 16 + l15;
          if (kc > qq) s = -1e30f;
        }
        sv[j][r] = s;
      }
    }

    // online softmax, write P (bf16) to per-wave LDS
#pragma unroll
    for (int r = 0; r < 8; ++r) {
      float t = fmaxf(fmaxf(sv[0][r], sv[1][r]), fmaxf(sv[2][r], sv[3][r]));
      t = hmax16(t);
      float mnew = fmaxf(mrow[r], t);
      float corr = __expf(mrow[r] - mnew);
      float rsum = 0.0f;
#pragma unroll
      for (int j = 0; j < 4; ++j) {
        float pexp = __expf(sv[j][r] - mnew);
        sP[wbase + (r + rowAdd) * 64 + j * 16 + l15] = f32_to_bf16(pexp);
        rsum += pexp;
      }
      rsum = hsum16(rsum);
      lrow[r] = lrow[r] * corr + rsum;
      mrow[r] = mnew;
#pragma unroll
      for (int j = 0; j < 4; ++j) oacc[j][r] = oacc[j][r] * corr;
    }

    // O += P @ V   (16 q x 64 depth, K = 64 keys)
#pragma unroll
    for (int kk = 0; kk < 2; ++kk) {
      v16bf ap = *(const v16bf*)(&sP[wbase + l15 * 64 + kk * 32 + half16]);
#pragma unroll
      for (int j = 0; j < 4; ++j) {
        v16bf bv = *(const v16bf*)(&cVt[(j * 16 + l15) * ATT_LDS + kk * 32 + half16]);
        oacc[j] = wmma_bf16(ap, bv, oacc[j]);
      }
    }

    wait_async();      // own prefetch done
    __syncthreads();   // all waves: prefetch visible + done with old buffer
    buf ^= 1;
  }

#pragma unroll
  for (int j = 0; j < 4; ++j)
#pragma unroll
    for (int r = 0; r < 8; ++r) {
      int qrow = q0 + wave * 16 + r + rowAdd;
      int col  = hh * DEPTH_ + j * 16 + l15;
      float ov = oacc[j][r] / lrow[r];
      O[(size_t)(bb * Sq + qrow) * D_ + col] = f32_to_bf16(ov);
    }
}

// ------------------------------------------------- residual add + layernorm
__launch_bounds__(256)
__global__ void add_layernorm(const float* __restrict__ X, const float* __restrict__ R,
                              const float* __restrict__ g, const float* __restrict__ b,
                              float* __restrict__ Yf, unsigned short* __restrict__ Yb) {
  const int row = blockIdx.x;
  const int tid = threadIdx.x;
  const float* x = X + (size_t)row * D_;
  const float* r = R + (size_t)row * D_;
  __shared__ float red[8];

  float vals[2], s = 0.0f;
#pragma unroll
  for (int i = 0; i < 2; ++i) {
    vals[i] = x[tid + 256 * i] + r[tid + 256 * i];
    s += vals[i];
  }
  for (int off = 16; off > 0; off >>= 1) s += __shfl_xor(s, off, 32);
  if ((tid & 31) == 0) red[tid >> 5] = s;
  __syncthreads();
  float tot = 0.0f;
#pragma unroll
  for (int i = 0; i < 8; ++i) tot += red[i];
  const float mean = tot / (float)D_;
  __syncthreads();

  float v = 0.0f;
#pragma unroll
  for (int i = 0; i < 2; ++i) { float d = vals[i] - mean; v += d * d; }
  for (int off = 16; off > 0; off >>= 1) v += __shfl_xor(v, off, 32);
  if ((tid & 31) == 0) red[tid >> 5] = v;
  __syncthreads();
  float vt = 0.0f;
#pragma unroll
  for (int i = 0; i < 8; ++i) vt += red[i];
  const float inv = rsqrtf(vt / (float)D_ + 1e-6f);

#pragma unroll
  for (int i = 0; i < 2; ++i) {
    int c = tid + 256 * i;
    float y = (vals[i] - mean) * inv * g[c] + b[c];
    if (Yf) Yf[(size_t)row * D_ + c] = y;
    if (Yb) Yb[(size_t)row * D_ + c] = f32_to_bf16(y);
  }
}

// ---------------------------------------------------------------- launcher
extern "C" void kernel_launch(void* const* d_in, const int* in_sizes, int n_in,
                              void* d_out, int out_size, void* d_ws, size_t ws_size,
                              hipStream_t stream) {
  (void)in_sizes; (void)n_in; (void)out_size; (void)ws_size;
  const float* x   = (const float*)d_in[0];
  const float* enc = (const float*)d_in[1];
  const float* pad = (const float*)d_in[3];
  const float* w1q = (const float*)d_in[4];  const float* b1q = (const float*)d_in[5];
  const float* w1k = (const float*)d_in[6];  const float* b1k = (const float*)d_in[7];
  const float* w1v = (const float*)d_in[8];  const float* b1v = (const float*)d_in[9];
  const float* w1o = (const float*)d_in[10]; const float* b1o = (const float*)d_in[11];
  const float* w2q = (const float*)d_in[12]; const float* b2q = (const float*)d_in[13];
  const float* w2k = (const float*)d_in[14]; const float* b2k = (const float*)d_in[15];
  const float* w2v = (const float*)d_in[16]; const float* b2v = (const float*)d_in[17];
  const float* w2o = (const float*)d_in[18]; const float* b2o = (const float*)d_in[19];
  const float* fw1 = (const float*)d_in[20]; const float* fb1 = (const float*)d_in[21];
  const float* fw2 = (const float*)d_in[22]; const float* fb2 = (const float*)d_in[23];
  const float* g1  = (const float*)d_in[24]; const float* be1 = (const float*)d_in[25];
  const float* g2  = (const float*)d_in[26]; const float* be2 = (const float*)d_in[27];
  const float* g3  = (const float*)d_in[28]; const float* be3 = (const float*)d_in[29];

  const size_t NTOK = (size_t)B_ * S_;
  const size_t NACT = NTOK * D_;

  char* p = (char*)d_ws;
  auto alloc = [&](size_t bytes) -> char* {
    char* r = p; p += (bytes + 255) & ~(size_t)255; return r;
  };
  unsigned short* bfAct  = (unsigned short*)alloc(NACT * 2);
  unsigned short* bfE    = (unsigned short*)alloc(NACT * 2);
  unsigned short* bfW    = (unsigned short*)alloc((size_t)4194304 * 2);
  unsigned short* Qbf    = (unsigned short*)alloc(NACT * 2);
  unsigned short* Kbf    = (unsigned short*)alloc(NACT * 2);
  unsigned short* Vbf    = (unsigned short*)alloc(NACT * 2);
  unsigned short* bfCore = (unsigned short*)alloc(NACT * 2);
  unsigned short* Hbf    = (unsigned short*)alloc(NTOK * DFF_ * 2);
  float* F1    = (float*)alloc(NACT * 4);
  float* out1f = (float*)alloc(NACT * 4);
  float* out2f = (float*)alloc(NACT * 4);

  unsigned short* cw1q = bfW + 0;        unsigned short* cw1k = bfW + 262144;
  unsigned short* cw1v = bfW + 524288;   unsigned short* cw1o = bfW + 786432;
  unsigned short* cw2q = bfW + 1048576;  unsigned short* cw2k = bfW + 1310720;
  unsigned short* cw2v = bfW + 1572864;  unsigned short* cw2o = bfW + 1835008;
  unsigned short* cfw1 = bfW + 2097152;  unsigned short* cfw2 = bfW + 3145728;

  auto cvt = [&](const float* src, unsigned short* dst, int n) {
    int blocks = (n + 1023) >> 10;
    hipLaunchKernelGGL(cvt_f32_to_bf16, dim3(blocks), dim3(256), 0, stream, src, dst, n);
  };
  auto cvtT = [&](const float* src, unsigned short* dst, int K, int N) {
    hipLaunchKernelGGL(cvt_transpose_bf16, dim3(N / 32, K / 32), dim3(256), 0, stream,
                       src, dst, K, N);
  };
  auto gemm_f32 = [&](const unsigned short* A, const unsigned short* W, const float* bias,
                      float* Cf, int M, int N, int K) {
    hipLaunchKernelGGL(HIP_KERNEL_NAME(gemm_bf16_wmma<0>), dim3(N / BN, M / BM), dim3(128),
                       0, stream, A, W, bias, Cf, (unsigned short*)nullptr, M, N, K);
  };
  auto gemm_bf = [&](const unsigned short* A, const unsigned short* W, const float* bias,
                     unsigned short* Cb, int M, int N, int K) {
    hipLaunchKernelGGL(HIP_KERNEL_NAME(gemm_bf16_wmma<1>), dim3(N / BN, M / BM), dim3(128),
                       0, stream, A, W, bias, (float*)nullptr, Cb, M, N, K);
  };
  auto gemm_relu = [&](const unsigned short* A, const unsigned short* W, const float* bias,
                       unsigned short* Cb, int M, int N, int K) {
    hipLaunchKernelGGL(HIP_KERNEL_NAME(gemm_bf16_wmma<2>), dim3(N / BN, M / BM), dim3(128),
                       0, stream, A, W, bias, (float*)nullptr, Cb, M, N, K);
  };
  auto ln = [&](const float* X, const float* R, const float* g, const float* b,
                float* Yf, unsigned short* Yb) {
    hipLaunchKernelGGL(add_layernorm, dim3((int)NTOK), dim3(256), 0, stream,
                       X, R, g, b, Yf, Yb);
  };

  // 0) precision conversion (+ weight transpose to Wt[N][K])
  cvt(x,   bfAct, (int)NACT);
  cvt(enc, bfE,   (int)NACT);
  cvtT(w1q, cw1q, D_, D_);   cvtT(w1k, cw1k, D_, D_);
  cvtT(w1v, cw1v, D_, D_);   cvtT(w1o, cw1o, D_, D_);
  cvtT(w2q, cw2q, D_, D_);   cvtT(w2k, cw2k, D_, D_);
  cvtT(w2v, cw2v, D_, D_);   cvtT(w2o, cw2o, D_, D_);
  cvtT(fw1, cfw1, D_, DFF_); cvtT(fw2, cfw2, DFF_, D_);

  // 1) masked self-attention
  gemm_bf(bfAct, cw1q, b1q, Qbf, (int)NTOK, D_, D_);
  gemm_bf(bfAct, cw1k, b1k, Kbf, (int)NTOK, D_, D_);
  gemm_bf(bfAct, cw1v, b1v, Vbf, (int)NTOK, D_, D_);
  hipLaunchKernelGGL(flash_attn_wmma, dim3(S_ / 64, H_, B_), dim3(128), 0, stream,
                     Qbf, Kbf, Vbf, nullptr, bfCore, S_, S_, 1, 0);
  gemm_f32(bfCore, cw1o, b1o, F1, (int)NTOK, D_, D_);
  ln(F1, x, g1, be1, out1f, bfAct);

  // 2) cross-attention (q = out1, k/v = enc)
  gemm_bf(bfAct, cw2q, b2q, Qbf, (int)NTOK, D_, D_);
  gemm_bf(bfE,   cw2k, b2k, Kbf, (int)NTOK, D_, D_);
  gemm_bf(bfE,   cw2v, b2v, Vbf, (int)NTOK, D_, D_);
  hipLaunchKernelGGL(flash_attn_wmma, dim3(S_ / 64, H_, B_), dim3(128), 0, stream,
                     Qbf, Kbf, Vbf, pad, bfCore, S_, SE_, 0, 1);
  gemm_f32(bfCore, cw2o, b2o, F1, (int)NTOK, D_, D_);
  ln(F1, out1f, g2, be2, out2f, bfE);

  // 3) FFN
  gemm_relu(bfE, cfw1, fb1, Hbf, (int)NTOK, DFF_, D_);
  gemm_f32(Hbf, cfw2, fb2, F1, (int)NTOK, D_, DFF_);
  ln(F1, out2f, g3, be3, (float*)d_out, nullptr);
}
